// Winograd2d_22677427323669
// MI455X (gfx1250) — matrix operand order repaired and checked by reading, not compiled, not run on hardware
//
#include <hip/hip_runtime.h>
#include <hip/hip_bf16.h>

// ---------------------------------------------------------------------------
// Winograd F(4x4,3x3) conv, N=32,C=256,H=W=56,O=256 on gfx1250 (MI455X)
// Stage 1: weight transform -> U[tu][o][c] (bf16)
// Stage 2: input  transform -> V[tu][p][c] (bf16)
// Stage 3: FUSED 36x GEMM + inverse transform + bias -> y
//          One wave owns a 16(o) x 16(p) tile. Per (t,u): K=256 GEMM as two
//          interleaved split-K v_wmma_f32_16x16x32_bf16 chains; the finished
//          m[t][u] folds lane-locally into the inverse transform (C/D layout
//          identical across accumulators -> no LDS, no M buffer).
//          t-loop kept rolled to bound VGPR live ranges (<256, no spills).
// ---------------------------------------------------------------------------

#define Nn   32
#define Cc   256
#define Oc   256
#define Hh   56
#define Ww   56
#define nT   14
#define Pt   6272          // 32 * 14 * 14
#define TU   36

#define U_BYTES  ((size_t)TU * Oc * Cc * 2)              //   4,718,592
#define V_BYTES  ((size_t)TU * Pt * Cc * 2)              // 115,605,504

typedef __attribute__((ext_vector_type(16))) __bf16 v16bf;
typedef __attribute__((ext_vector_type(8)))  __bf16 v8bf;
typedef __attribute__((ext_vector_type(8)))  float  v8f;

__device__ constexpr float BTm[6][6] = {
    { 4,  0, -5,  0, 1, 0},
    { 0, -4, -4,  1, 1, 0},
    { 0,  4, -4, -1, 1, 0},
    { 0, -2, -1,  2, 1, 0},
    { 0,  2, -1, -2, 1, 0},
    { 0,  4,  0, -5, 0, 1}};
__device__ constexpr float Gm[6][3] = {
    { 0.25f,     0.f,        0.f},
    {-1.f/6.f,  -1.f/6.f,   -1.f/6.f},
    {-1.f/6.f,   1.f/6.f,   -1.f/6.f},
    { 1.f/24.f,  1.f/12.f,   1.f/6.f},
    { 1.f/24.f, -1.f/12.f,   1.f/6.f},
    { 0.f,       0.f,        1.f}};
__device__ constexpr float ATm[4][6] = {
    {1, 1,  1, 1,  1, 0},
    {0, 1, -1, 2, -2, 0},
    {0, 1,  1, 4,  4, 0},
    {0, 1, -1, 8, -8, 1}};

// ---------------------------------------------------------------------------
// Stage 1: weight transform, one thread per (o,c)
// ---------------------------------------------------------------------------
__global__ __launch_bounds__(256)
void wg_transform_kernel(const float* __restrict__ w, const float* __restrict__ mask,
                         __hip_bfloat16* __restrict__ U)
{
    int idx = blockIdx.x * blockDim.x + threadIdx.x;
    if (idx >= Oc * Cc) return;
    const int o = idx / Cc, c = idx % Cc;

    float g[3][3];
    const float* wp = w + (size_t)idx * 9;
#pragma unroll
    for (int i = 0; i < 3; ++i)
#pragma unroll
        for (int j = 0; j < 3; ++j) g[i][j] = wp[i * 3 + j];

    float tmp[6][3];
#pragma unroll
    for (int t = 0; t < 6; ++t)
#pragma unroll
        for (int j = 0; j < 3; ++j)
            tmp[t][j] = Gm[t][0] * g[0][j] + Gm[t][1] * g[1][j] + Gm[t][2] * g[2][j];

    const float* mp = mask + (size_t)idx * 36;
#pragma unroll
    for (int t = 0; t < 6; ++t)
#pragma unroll
        for (int u = 0; u < 6; ++u) {
            float v = tmp[t][0] * Gm[u][0] + tmp[t][1] * Gm[u][1] + tmp[t][2] * Gm[u][2];
            v *= (1.0f - mp[t * 6 + u]);
            U[((size_t)(t * 6 + u) * Oc + o) * Cc + c] = __float2bfloat16(v);
        }
}

// ---------------------------------------------------------------------------
// Stage 2: input transform. 256 threads over c; blockIdx.x = th*14+tw,
// blockIdx.y = n. Writes V[tu][p][c] (K contiguous) -> coalesced per tu.
// ---------------------------------------------------------------------------
__global__ __launch_bounds__(256)
void in_transform_kernel(const float* __restrict__ x, __hip_bfloat16* __restrict__ V)
{
    const int c  = threadIdx.x;
    const int n  = blockIdx.y;
    const int th = blockIdx.x / nT, tw = blockIdx.x % nT;

    const float* xp = x + ((size_t)n * Cc + c) * (Hh * Ww);
    const int h0 = th * 4 - 1, w0 = tw * 4 - 1;

    float d[6][6];
#pragma unroll
    for (int i = 0; i < 6; ++i) {
        const int h = h0 + i;
        const bool hv = (h >= 0) && (h < Hh);
#pragma unroll
        for (int j = 0; j < 6; ++j) {
            const int wcol = w0 + j;
            const bool ok = hv && (wcol >= 0) && (wcol < Ww);
            d[i][j] = ok ? xp[h * Ww + wcol] : 0.0f;
        }
    }

    float tmp[6][6];  // BT * d
#pragma unroll
    for (int t = 0; t < 6; ++t)
#pragma unroll
        for (int j = 0; j < 6; ++j) {
            float s = 0.f;
#pragma unroll
            for (int i = 0; i < 6; ++i) s += BTm[t][i] * d[i][j];
            tmp[t][j] = s;
        }

    const int p = n * (nT * nT) + th * nT + tw;
#pragma unroll
    for (int t = 0; t < 6; ++t)
#pragma unroll
        for (int u = 0; u < 6; ++u) {
            float s = 0.f;
#pragma unroll
            for (int j = 0; j < 6; ++j) s += tmp[t][j] * BTm[u][j];
            V[((size_t)(t * 6 + u) * Pt + p) * Cc + c] = __float2bfloat16(s);
        }
}

// ---------------------------------------------------------------------------
// fragment load helpers (ISA 7.12.2 layouts)
// ---------------------------------------------------------------------------
__device__ __forceinline__ v16bf load_fragA(const __bf16* p)
{
    v8bf lo = *(const v8bf*)(p);        // halves K..K+7   (lane sel adds +8)
    v8bf hi = *(const v8bf*)(p + 16);   // halves K+16..K+23
    v16bf a;
#pragma unroll
    for (int e = 0; e < 8; ++e) { a[e] = lo[e]; a[e + 8] = hi[e]; }
    return a;
}
__device__ __forceinline__ v16bf load_fragB(const __bf16* p)
{
    v8bf lo = *(const v8bf*)(p);        // halves K..K+7   (lane sel adds +8)
    v8bf hi = *(const v8bf*)(p + 16);   // halves K+16..K+23
    v16bf b;
#pragma unroll
    for (int e = 0; e < 8; ++e) { b[e] = lo[e]; b[e + 8] = hi[e]; }
    return b;
}

// ---------------------------------------------------------------------------
// Stage 3: fused GEMM + inverse transform.
// Block = 256 threads = 8 waves. gw: o_blk = gw & 15, p_blk = gw >> 4.
// ---------------------------------------------------------------------------
__global__ __launch_bounds__(256)
void fused_gemm_out_kernel(const __bf16* __restrict__ U, const __bf16* __restrict__ V,
                           const float* __restrict__ bias, float* __restrict__ y)
{
    const int lane  = threadIdx.x & 31;
    const int gwave = blockIdx.x * 8 + (threadIdx.x >> 5);
    const int o_blk = gwave & 15;          // 16 blocks of 16 output channels
    const int p_blk = gwave >> 4;          // 392 blocks of 16 tiles

    const int rowA0 = o_blk * 16 + (lane & 15);
    const int colB0 = p_blk * 16 + (lane & 15);
    const int aSel  = (lane >> 4) ? 8 : 0;    // A frag: lanes 16-31 hold K+8/K+24
    const int bSel  = (lane >> 4) ? 8 : 0;    // B frag: lanes 16-31 hold K+8/K+24

    const __bf16* Arow = U + (size_t)rowA0 * Cc + aSel;   // + tu*Oc*Cc + kk
    const __bf16* Bcol = V + (size_t)colB0 * Cc + bSel;   // + tu*Pt*Cc + kk

    v8f yacc[4][4];               // final 4x4 spatial outputs per (o,p) element
#pragma unroll
    for (int r = 0; r < 4; ++r)
#pragma unroll
        for (int q = 0; q < 4; ++q) yacc[r][q] = v8f{};

#pragma unroll
    for (int u = 0; u < 6; ++u) {
        v8f tmpR[4];              // sum_t AT[r][t] * m[t][u]
#pragma unroll
        for (int r = 0; r < 4; ++r) tmpR[r] = v8f{};

        // t-loop kept ROLLED: bounds VGPR live ranges (no cross-iteration
        // load hoisting), keeps allocation under 256 VGPRs.
#pragma unroll 1
        for (int t = 0; t < 6; ++t) {
            const int tu = t * 6 + u;
            const __bf16* pa = Arow + (size_t)tu * (Oc * Cc);
            const __bf16* pb = Bcol + (size_t)tu * ((size_t)Pt * Cc);

            // prefetch next (t,u) panels (global_prefetch_b8)
            {
                const int tun = (t < 5) ? (tu + 6) : (u < 5 ? u + 1 : tu);
                __builtin_prefetch(Arow + (size_t)tun * (Oc * Cc), 0, 1);
                __builtin_prefetch(Bcol + (size_t)tun * ((size_t)Pt * Cc), 0, 1);
            }

            // split-K: two interleaved WMMA chains, fragments reloaded right
            // after consumption (4 fragment buffers live, 32 VGPRs).
            v8f acc0 = v8f{}, acc1 = v8f{};
            v16bf a0 = load_fragA(pa);
            v16bf b0 = load_fragB(pb);
            v16bf a1 = load_fragA(pa + 32);
            v16bf b1 = load_fragB(pb + 32);

#pragma unroll
            for (int s = 0; s < 4; ++s) {
                acc0 = __builtin_amdgcn_wmma_f32_16x16x32_bf16(
                    false, a0, false, b0, (short)0, acc0, false, false);
                if (s < 3) {
                    a0 = load_fragA(pa + 64 * (s + 1));
                    b0 = load_fragB(pb + 64 * (s + 1));
                }
                acc1 = __builtin_amdgcn_wmma_f32_16x16x32_bf16(
                    false, a1, false, b1, (short)0, acc1, false, false);
                if (s < 3) {
                    a1 = load_fragA(pa + 64 * (s + 1) + 32);
                    b1 = load_fragB(pb + 64 * (s + 1) + 32);
                }
            }

            // m[t][u] = acc0 + acc1; fold with (uniform) runtime coefficients
            const float cf0 = ATm[0][t];
            const float cf1 = ATm[1][t];
            const float cf2 = ATm[2][t];
            const float cf3 = ATm[3][t];
#pragma unroll
            for (int e = 0; e < 8; ++e) {
                const float m = acc0[e] + acc1[e];
                tmpR[0][e] += cf0 * m;
                tmpR[1][e] += cf1 * m;
                tmpR[2][e] += cf2 * m;
                tmpR[3][e] += cf3 * m;
            }
        }

        // fold tmpR into yacc with compile-time sparse AT coefficients
#pragma unroll
        for (int q = 0; q < 4; ++q) {
            const float cf = ATm[q][u];
            if (cf != 0.f) {
#pragma unroll
                for (int r = 0; r < 4; ++r)
#pragma unroll
                    for (int e = 0; e < 8; ++e) yacc[r][q][e] += cf * tmpR[r][e];
            }
        }
    }

    // ---- epilogue: bias + direct store of 4x4 spatial block per (o,p) ----
    // C/D layout: slot e, lane l -> o = o_blk*16 + e + 8*(l>=16), p = colB0
    const int p  = colB0;
    const int n  = p / (nT * nT);
    const int q2 = p % (nT * nT);
    const int th = q2 / nT, tw = q2 % nT;
    const int rsel = (lane >> 4) ? 8 : 0;

#pragma unroll
    for (int e = 0; e < 8; ++e) {
        const int o = o_blk * 16 + rsel + e;
        const float b = bias[o];
        float* yp = y + ((size_t)(n * Oc + o) * Hh + th * 4) * Ww + tw * 4;
#pragma unroll
        for (int r = 0; r < 4; ++r) {
            float4 v4;
            v4.x = yacc[r][0][e] + b;
            v4.y = yacc[r][1][e] + b;
            v4.z = yacc[r][2][e] + b;
            v4.w = yacc[r][3][e] + b;
            *(float4*)(yp + r * Ww) = v4;
        }
    }
}

// ---------------------------------------------------------------------------
extern "C" void kernel_launch(void* const* d_in, const int* in_sizes, int n_in,
                              void* d_out, int out_size, void* d_ws, size_t ws_size,
                              hipStream_t stream)
{
    const float* x      = (const float*)d_in[0];
    const float* weight = (const float*)d_in[1];
    const float* bias   = (const float*)d_in[2];
    const float* mask   = (const float*)d_in[3];
    float*       y      = (float*)d_out;

    char* ws = (char*)d_ws;
    __hip_bfloat16* U = (__hip_bfloat16*)(ws);
    __hip_bfloat16* V = (__hip_bfloat16*)(ws + U_BYTES);

    // Stage 1: 65536 (o,c) pairs
    wg_transform_kernel<<<dim3((Oc * Cc + 255) / 256), 256, 0, stream>>>(weight, mask, U);
    // Stage 2: one block per (tile, image), 256 channels per block
    in_transform_kernel<<<dim3(nT * nT, Nn), 256, 0, stream>>>(x, V);
    // Stage 3: fused GEMM + inverse transform + bias.
    // waves = 16 o_blks * 392 p_blks = 6272 -> 784 blocks of 8 waves.
    fused_gemm_out_kernel<<<dim3((16 * (Pt / 16)) / 8), 256, 0, stream>>>(
        (const __bf16*)U, (const __bf16*)V, bias, y);
}
